// MyConv2d_4724464026153
// MI455X (gfx1250) — matrix operand order repaired and checked by reading, not compiled, not run on hardware
//
#include <hip/hip_runtime.h>
#include <stdint.h>

// ---------------------------------------------------------------------------
// Conv2d 3x3 (C_in=128 -> C_out=256, H=W=256, pad=1, stride=1) + bias, fp32.
//
// Strategy (MI455X / gfx1250):
//  * Compute-bound (AI ~380 FLOP/B): implicit GEMM on v_wmma_f32_16x16x32_bf16
//    using split-bf16 (x = xh + xl, w = wh + wl; keep xh*wh + xh*wl + xl*wh).
//    3 bf16 WMMAs per 32-K vs 8 fp32 WMMAs per 32-K -> ~2.7x fewer matrix ops
//    at ~2^-16 relative error (near-fp32).
//  * One-time prepack kernels split/pack x and w into d_ws in the exact LDS
//    image layout; split x (33.5 MB) + w (1.2 MB) fit in the 192 MB L2.
//  * x staging: global_load_async_to_lds_b128 (ASYNCcnt), zero-pad borders.
//  * w staging: one TDM tensor_load_to_lds per chunk (74 KB contiguous,
//    TENSORcnt), issued by wave 0 only; async-copy fallback.
// ---------------------------------------------------------------------------

typedef __attribute__((ext_vector_type(16))) __bf16        v16bf;
typedef __attribute__((ext_vector_type(8)))  float         v8f;
typedef __attribute__((ext_vector_type(8)))  unsigned int  v8u;

#define C_IN   128
#define C_OUT  256
#define HH     256
#define WW     256

#define CO_TILE   64
#define PX_TILE   64
#define CI_CHUNK  32
#define NPAIR     (CI_CHUNK / 2)        // 16 packed bf16 pairs per position
#define NPAIR_ALL (C_IN / 2)            // 64 pairs total
#define X_DWORDS  (3 * 66 * NPAIR)      // LDS x image: 3 rows x 66 cols x 16 pairs
#define W_DWORDS  (CO_TILE * 9 * NPAIR) // LDS w image: 64 co x 9 taps x 16 pairs
#define SMEM_DW   (2 * X_DWORDS + 2 * W_DWORDS)

// d_ws layout (dwords):
//   xhi: [256 rows][256 cols][64 pairs]                  = 4,194,304 dw
//   xlo: same                                            = 4,194,304 dw
//   w:   [4 co-tiles][4 chunks][hi W_DWORDS | lo W_DWORDS] = 294,912 dw
#define WS_XHI   0
#define WS_XLO   (WS_XHI + (size_t)HH * WW * NPAIR_ALL)
#define WS_W     (WS_XLO + (size_t)HH * WW * NPAIR_ALL)

// ---- CDNA5 async global->LDS copy (ASYNCcnt) with sync fallback -----------
#if defined(__HIP_DEVICE_COMPILE__) &&                                     \
    __has_builtin(__builtin_amdgcn_global_load_async_to_lds_b128) &&       \
    __has_builtin(__builtin_amdgcn_s_wait_asynccnt)
#define USE_ASYNC_LDS 1
typedef int v4i __attribute__((vector_size(16)));
typedef __attribute__((address_space(1))) v4i* gvec_p;   // global (prints __device__)
typedef __attribute__((address_space(3))) v4i* lvec_p;   // LDS (32-bit ptr)
__device__ __forceinline__ void async_cp16(void* lds_dst, const void* gsrc) {
  // Generic LDS addresses carry the LDS byte offset in the low 32 bits.
  __builtin_amdgcn_global_load_async_to_lds_b128(
      (gvec_p)(uintptr_t)gsrc,
      (lvec_p)(unsigned int)(uintptr_t)lds_dst, 0, 0);
}
__device__ __forceinline__ void async_wait0() {
  __builtin_amdgcn_s_wait_asynccnt(0);
}
#else
#define USE_ASYNC_LDS 0
__device__ __forceinline__ void async_cp16(void* lds_dst, const void* gsrc) {
  *(uint4*)lds_dst = *(const uint4*)gsrc;
}
__device__ __forceinline__ void async_wait0() {}
#endif

// ---- CDNA5 Tensor Data Mover: 1-D global->LDS DMA (TENSORcnt) -------------
#if defined(__HIP_DEVICE_COMPILE__) &&                                     \
    __has_builtin(__builtin_amdgcn_tensor_load_to_lds) &&                  \
    __has_builtin(__builtin_amdgcn_s_wait_tensorcnt)
#define USE_TDM 1
typedef unsigned int u32x4 __attribute__((ext_vector_type(4)));
typedef int          i32x4 __attribute__((ext_vector_type(4)));
typedef int          i32x8 __attribute__((ext_vector_type(8)));
// Copy n_dwords contiguous dwords from global to LDS via one TDM descriptor.
__device__ __forceinline__ void tdm_copy_1d(void* lds_dst, const void* gsrc,
                                            unsigned int n_dwords) {
  unsigned long long ga = (unsigned long long)(uintptr_t)gsrc;
  u32x4 g0;
  g0.x = 1u;                                        // count=1, user mode
  g0.y = (unsigned int)(uintptr_t)lds_dst;          // lds_addr (bytes)
  g0.z = (unsigned int)ga;                          // global_addr[31:0]
  g0.w = ((unsigned int)(ga >> 32) & 0x01FFFFFFu)   // global_addr[56:32]
         | (2u << 30);                              // type=2 ("image")
  i32x8 g1;
  g1[0] = (int)(2u << 16);                          // data_size=4B; no flags
  g1[1] = (int)((n_dwords & 0xFFFFu) << 16);        // tensor_dim0[15:0]
  g1[2] = (int)((n_dwords >> 16) | (1u << 16));     // tensor_dim0[31:16]; dim1=1
  g1[3] = (int)((n_dwords & 0xFFFFu) << 16);        // tile_dim0 = n_dwords
  g1[4] = 0;                                        // tile_dim1/2 unused
  g1[5] = (int)n_dwords;                            // dim0_stride[31:0]
  g1[6] = 0;
  g1[7] = 0;
  i32x4 gz = {0, 0, 0, 0};
#if __clang_major__ >= 23
  i32x8 gz8 = {0, 0, 0, 0, 0, 0, 0, 0};
  __builtin_amdgcn_tensor_load_to_lds(g0, g1, gz, gz, gz8, 0);
#else
  __builtin_amdgcn_tensor_load_to_lds(g0, g1, gz, gz, 0);
#endif
}
__device__ __forceinline__ void tdm_wait0() {
  __builtin_amdgcn_s_wait_tensorcnt(0);
}
#else
#define USE_TDM 0
__device__ __forceinline__ void tdm_wait0() {}
#endif

// Split fp32 -> bf16 hi (truncate) + exact residual lo (truncate).
__device__ __forceinline__ void split2(float x, unsigned int& hi16,
                                       unsigned int& lo16) {
  unsigned int u = __float_as_uint(x);
  unsigned int h = u & 0xffff0000u;
  float lf = x - __uint_as_float(h);          // exact
  hi16 = h >> 16;
  lo16 = __float_as_uint(lf) >> 16;
}

__device__ __forceinline__ unsigned int split_pack(float x0, float x1,
                                                   unsigned int& lo_pack) {
  unsigned int h0, l0, h1, l1;
  split2(x0, h0, l0);
  split2(x1, h1, l1);
  lo_pack = l0 | (l1 << 16);
  return h0 | (h1 << 16);
}

__device__ __forceinline__ v16bf load_frag(const unsigned int* base,
                                           int off0, int off1) {
  uint4 a0 = *(const uint4*)(base + off0);
  uint4 a1 = *(const uint4*)(base + off1);
  v8u u = {a0.x, a0.y, a0.z, a0.w, a1.x, a1.y, a1.z, a1.w};
  return __builtin_bit_cast(v16bf, u);
}

// ---------------------------------------------------------------------------
// Prepack weights: [co-tile][chunk][hi image | lo image] — exact LDS image,
// hi and lo contiguous so one TDM op stages both.
// ---------------------------------------------------------------------------
extern "C" __global__ __launch_bounds__(256)
void prepack_w(const float* __restrict__ Wt, unsigned int* __restrict__ ws) {
  int idx = blockIdx.x * 256 + threadIdx.x;     // [0, 4*4*W_DWORDS)
  int blk   = idx / W_DWORDS;                   // ct*4 + chunk
  int j     = idx % W_DWORDS;
  int ct    = blk >> 2;
  int chunk = blk & 3;
  int p   = j & 15;
  int t   = j >> 4;
  int tap = t % 9;
  int cr  = t / 9;
  int co  = ct * CO_TILE + cr;
  int ci  = chunk * CI_CHUNK + 2 * p;
  const float* wp = Wt + (size_t)co * (C_IN * 9) + ci * 9 + tap;
  unsigned int lo;
  unsigned int hi = split_pack(wp[0], wp[9], lo);
  unsigned int* dst = ws + WS_W + (size_t)blk * (2 * W_DWORDS);
  dst[j]            = hi;
  dst[W_DWORDS + j] = lo;
}

// ---------------------------------------------------------------------------
// Prepack x: split + pack to [row][col][pair] (pair innermost), transposing
// through LDS so both the reads (col-fast) and writes (pair-fast) coalesce.
// One block per (row, 64-col strip).
// ---------------------------------------------------------------------------
#define XP_PAD 129                       // [64 cols][129] shorts, padded
extern "C" __global__ __launch_bounds__(256)
void prepack_x(const float* __restrict__ X, unsigned int* __restrict__ ws) {
  __shared__ unsigned short sh_hi[64 * XP_PAD];
  __shared__ unsigned short sh_lo[64 * XP_PAD];
  unsigned int* xhi = ws + WS_XHI;
  unsigned int* xlo = ws + WS_XLO;

  const int tid  = threadIdx.x;
  const int row  = blockIdx.x >> 2;
  const int col0 = (blockIdx.x & 3) * 64;

  for (int i = tid; i < C_IN * 64; i += 256) {
    int c  = i & 63;
    int ci = i >> 6;
    float v = X[(size_t)ci * (HH * WW) + row * WW + col0 + c];
    unsigned int h, l;
    split2(v, h, l);
    sh_hi[c * XP_PAD + ci] = (unsigned short)h;
    sh_lo[c * XP_PAD + ci] = (unsigned short)l;
  }
  __syncthreads();

  for (int i = tid; i < 64 * NPAIR_ALL; i += 256) {
    int p = i & 63;
    int c = i >> 6;
    unsigned int hi = (unsigned int)sh_hi[c * XP_PAD + 2 * p] |
                      ((unsigned int)sh_hi[c * XP_PAD + 2 * p + 1] << 16);
    unsigned int lo = (unsigned int)sh_lo[c * XP_PAD + 2 * p] |
                      ((unsigned int)sh_lo[c * XP_PAD + 2 * p + 1] << 16);
    size_t dst = ((size_t)row * WW + col0 + c) * NPAIR_ALL + p;
    xhi[dst] = hi;
    xlo[dst] = lo;
  }
}

// ---------------------------------------------------------------------------
// Main kernel: 64 co x 64 px per block (8 wave32s), wave = 16 co x 32 px.
// ---------------------------------------------------------------------------
extern "C" __global__ __launch_bounds__(256)
void conv3x3_wmma_bf16x3(const unsigned int* __restrict__ ws,
                         const float* __restrict__ Bias,
                         float* __restrict__ Out) {
  extern __shared__ unsigned int smem[];
  unsigned int* xh = smem;                 // [3][66][NPAIR] hi pairs
  unsigned int* xl = xh + X_DWORDS;
  unsigned int* wh = xl + X_DWORDS;        // [64][9][NPAIR] hi pairs
  unsigned int* wl = wh + W_DWORDS;        // contiguous after wh

  const unsigned int* Xhi = ws + WS_XHI;
  const unsigned int* Xlo = ws + WS_XLO;
  const unsigned int* Wpk = ws + WS_W;

  const int tid  = threadIdx.x;
  const int lane = tid & 31;
  const int wid  = tid >> 5;
  const int half = lane >> 4;              // K-half select (lanes 0-15 / 16-31)
  const int n    = lane & 15;              // N (pixel) / M (co) within fragment
  const int cg   = wid & 3;                // wave's co sub-tile (16 rows each)
  const int g0   = (wid >> 2) * 32;        // wave's pixel base (0 or 32)

  const int h       = blockIdx.x >> 2;     // output row
  const int w0      = (blockIdx.x & 3) * PX_TILE;
  const int ct      = blockIdx.y;          // co tile
  const int co_base = ct * CO_TILE;

  v8f acc0 = {};
  v8f acc1 = {};

  for (int chunk = 0; chunk < C_IN / CI_CHUNK; ++chunk) {
    const int p0 = chunk * NPAIR;

    // ---- stage weight image: one TDM DMA for the whole 2*W_DWORDS run ----
    const unsigned int* wsrc = Wpk + (size_t)(ct * 4 + chunk) * (2 * W_DWORDS);
#if USE_TDM
    if (wid == 0)
      tdm_copy_1d(wh, wsrc, 2 * W_DWORDS);
#else
    for (int i = tid; i < (2 * W_DWORDS) / 4; i += 256)
      async_cp16(&wh[i * 4], wsrc + i * 4);
#endif

    // ---- stage x image: 3*66 segments of 16 pair-dwords (64B), hi+lo ----
    for (int i = tid; i < 3 * 66 * 4; i += 256) {
      int q  = i & 3;                      // 16B quarter of the 64B segment
      int t  = i >> 2;                     // r*66 + wc
      int wc = t % 66;
      int r  = t / 66;
      int row = h + r - 1;
      int col = w0 + wc - 1;
      unsigned int* dh = &xh[t * NPAIR + q * 4];
      unsigned int* dl = &xl[t * NPAIR + q * 4];
      if ((unsigned)row < HH && (unsigned)col < WW) {
        size_t src = ((size_t)row * WW + col) * NPAIR_ALL + p0 + q * 4;
        async_cp16(dh, Xhi + src);
        async_cp16(dl, Xlo + src);
      } else {
        uint4 z = {0u, 0u, 0u, 0u};
        *(uint4*)dh = z;
        *(uint4*)dl = z;
      }
    }
    async_wait0();            // own async transfers landed in LDS
    tdm_wait0();              // wave0's tensor DMA landed (no-op elsewhere)
    __syncthreads();          // everyone's landed

    // ---- 9 taps x (3 split products) x 2 pixel groups of 16x16x32 ----
#pragma unroll
    for (int tap = 0; tap < 9; ++tap) {
      const int kh = tap / 3;
      const int kw = tap % 3;

      // A fragment (w, 16 co x 32 k): pairs 8*half + v (contiguous)
      const int abase = ((cg * 16 + n) * 9 + tap) * NPAIR;
      v16bf Ah = load_frag(wh, abase + 8 * half, abase + 8 * half + 4);
      v16bf Al = load_frag(wl, abase + 8 * half, abase + 8 * half + 4);

      // B fragments (x, 32 k x 16 px): pairs 8*half + v (contiguous)
      const int bb0 = ((kh * 66) + (g0 + n + kw)) * NPAIR + 8 * half;
      v16bf B0h = load_frag(xh, bb0, bb0 + 4);
      v16bf B0l = load_frag(xl, bb0, bb0 + 4);
      acc0 = __builtin_amdgcn_wmma_f32_16x16x32_bf16(false, Ah, false, B0h,
                                                     (short)0, acc0, false, false);
      acc0 = __builtin_amdgcn_wmma_f32_16x16x32_bf16(false, Ah, false, B0l,
                                                     (short)0, acc0, false, false);
      acc0 = __builtin_amdgcn_wmma_f32_16x16x32_bf16(false, Al, false, B0h,
                                                     (short)0, acc0, false, false);

      const int bb1 = ((kh * 66) + (g0 + 16 + n + kw)) * NPAIR + 8 * half;
      v16bf B1h = load_frag(xh, bb1, bb1 + 4);
      v16bf B1l = load_frag(xl, bb1, bb1 + 4);
      acc1 = __builtin_amdgcn_wmma_f32_16x16x32_bf16(false, Ah, false, B1h,
                                                     (short)0, acc1, false, false);
      acc1 = __builtin_amdgcn_wmma_f32_16x16x32_bf16(false, Ah, false, B1l,
                                                     (short)0, acc1, false, false);
      acc1 = __builtin_amdgcn_wmma_f32_16x16x32_bf16(false, Al, false, B1h,
                                                     (short)0, acc1, false, false);
    }
    __syncthreads();
  }

  // ---- epilogue: bias + store (C/D layout: VGPR r -> M = r + 8*half) ----
#pragma unroll
  for (int gg = 0; gg < 2; ++gg) {
    v8f acc = gg ? acc1 : acc0;
    const int px = g0 + gg * 16 + n;
#pragma unroll
    for (int r = 0; r < 8; ++r) {
      const int co = co_base + cg * 16 + r + 8 * half;
      Out[(size_t)co * (HH * WW) + h * WW + (w0 + px)] = acc[r] + Bias[co];
    }
  }
}

extern "C" void kernel_launch(void* const* d_in, const int* in_sizes, int n_in,
                              void* d_out, int out_size, void* d_ws, size_t ws_size,
                              hipStream_t stream) {
  (void)in_sizes; (void)n_in; (void)out_size; (void)ws_size;
  const float* x  = (const float*)d_in[0];
  const float* w  = (const float*)d_in[1];
  const float* b  = (const float*)d_in[2];
  float* out      = (float*)d_out;
  unsigned int* ws = (unsigned int*)d_ws;   // needs ~34.7 MB scratch

  // 1) prepack weights: 4*4*W_DWORDS = 147456 dwords -> 576 blocks
  prepack_w<<<dim3(576), dim3(256), 0, stream>>>(w, ws);
  // 2) prepack x: one block per (row, 64-col strip)
  prepack_x<<<dim3(HH * 4), dim3(256), 0, stream>>>(x, ws);
  // 3) main conv
  dim3 grid(HH * (WW / PX_TILE), C_OUT / CO_TILE, 1);   // (1024, 4)
  size_t smem_bytes = (size_t)SMEM_DW * sizeof(unsigned int);  // ~97 KB
  conv3x3_wmma_bf16x3<<<grid, dim3(256), smem_bytes, stream>>>(ws, b, out);
}
